// AttentionNet_8048768712716
// MI455X (gfx1250) — matrix operand order repaired and checked by reading, not compiled, not run on hardware
//
#include <hip/hip_runtime.h>

typedef __attribute__((ext_vector_type(16))) __bf16 v16bf;
typedef __attribute__((ext_vector_type(8)))  float  v8f;

#define D_DIM   2048
#define H_DIM   512
#define ROWS    64
#define THREADS 512
#define NWAVES  16
#define HTILES  (H_DIM / 16)   // 32
#define NTILES  (ROWS / 16)    // 4

// ---------------------------------------------------------------------------
// Kernel 0: one-time f32 -> bf16 conversion of W1 (2 MB, stays L2-resident)
// ---------------------------------------------------------------------------
__global__ void cvt_w1_kernel(const float* __restrict__ W1, __bf16* __restrict__ W1bf, int n) {
    int i = blockIdx.x * 256 + threadIdx.x;
    if (i < n) W1bf[i] = (__bf16)W1[i];
}

// ---------------------------------------------------------------------------
// Kernel 1: fused scores = W2 . tanh(W1 x + b1) + b2 via bf16 WMMA
//   block = 64 x-rows, 512 threads (16 wave32s)
//   LDS: x tile as bf16 (64 x 2048 x 2B = 256KB) + 8KB partials
//   Register blocking: each wave owns 2 h-tiles x 4 n-tiles (8 accumulators)
//     -> W1 fragments reused 4x (L2 traffic /4), A fragments reused 2x,
//        and adjacent WMMAs hit distinct accumulators (no D->C RAW chains).
// ---------------------------------------------------------------------------
__global__ __launch_bounds__(THREADS)
void scores_kernel(const float* __restrict__ x, const __bf16* __restrict__ W1bf,
                   const float* __restrict__ b1, const float* __restrict__ W2,
                   const float* __restrict__ b2, float* __restrict__ scores, int N)
{
    __shared__ __bf16 xs[ROWS * D_DIM];
    __shared__ float  part[ROWS][HTILES];

    const int tid = threadIdx.x;
    const int n0  = blockIdx.x * ROWS;

    // Stage x tile -> LDS as bf16 (float2 loads, packed b32 LDS stores)
    for (int i = tid; i < ROWS * (D_DIM / 2); i += THREADS) {
        int r = i / (D_DIM / 2);
        int c = (i - r * (D_DIM / 2)) * 2;
        float2 v = make_float2(0.f, 0.f);
        int gr = n0 + r;
        if (gr < N) v = *(const float2*)(x + (size_t)gr * D_DIM + c);
        union { __bf16 h[2]; unsigned u; } pk;
        pk.h[0] = (__bf16)v.x;
        pk.h[1] = (__bf16)v.y;
        *(unsigned*)&xs[r * D_DIM + c] = pk.u;
    }
    __syncthreads();

    const int wave = tid >> 5;
    const int lane = tid & 31;
    const int lh   = lane & 15;   // row within lane-half
    const int hi   = lane >> 4;   // 0: low-K half, 1: high-K half

    const int ht0 = wave;         // first h-tile owned by this wave
    const int ht1 = wave + 16;    // second h-tile

    v8f c[8] = {};                // [g*4 + nt], g=0 -> ht0, g=1 -> ht1

    const __bf16* wbase0 = W1bf + (size_t)(ht0 * 16 + lh) * D_DIM;
    const __bf16* wbase1 = W1bf + (size_t)(ht1 * 16 + lh) * D_DIM;

    union BF { v16bf v; uint4 q[2]; };

    for (int k0 = 0; k0 < D_DIM; k0 += 32) {
        // A: 16x32 bf16; lane half hi holds K = k0+hi*8..+7 and k0+16+hi*8..+7 of row lh
        BF a[NTILES];
#pragma unroll
        for (int nt = 0; nt < NTILES; ++nt) {
            const __bf16* xr = &xs[(nt * 16 + lh) * D_DIM + k0 + hi * 8];
            a[nt].q[0] = *(const uint4*)xr;
            a[nt].q[1] = *(const uint4*)(xr + 16);
        }
        // B: 32x16 bf16; lane holds column n=lh => W1 row, K = k0+hi*8..+7 and k0+16+hi*8..+7
        BF b0, b1v;
        b0.q[0]  = *(const uint4*)(wbase0 + k0 + hi * 8);
        b0.q[1]  = *(const uint4*)(wbase0 + k0 + hi * 8 + 16);
        b1v.q[0] = *(const uint4*)(wbase1 + k0 + hi * 8);
        b1v.q[1] = *(const uint4*)(wbase1 + k0 + hi * 8 + 16);

#pragma unroll
        for (int nt = 0; nt < NTILES; ++nt)
            c[nt] = __builtin_amdgcn_wmma_f32_16x16x32_bf16(
                        false, a[nt].v, false, b0.v, (short)0, c[nt], false, false);
#pragma unroll
        for (int nt = 0; nt < NTILES; ++nt)
            c[4 + nt] = __builtin_amdgcn_wmma_f32_16x16x32_bf16(
                        false, a[nt].v, false, b1v.v, (short)0, c[4 + nt], false, false);
    }

    // Epilogue: tanh + W2 projection + in-tile h reduction
#pragma unroll
    for (int g = 0; g < 2; ++g) {
        const int ht  = (g == 0) ? ht0 : ht1;
        const float w2h = W2[ht * 16 + lh];
        const float b1h = b1[ht * 16 + lh];
#pragma unroll
        for (int nt = 0; nt < NTILES; ++nt) {
            float p[8];
#pragma unroll
            for (int e = 0; e < 8; ++e) p[e] = tanhf(c[g * 4 + nt][e] + b1h) * w2h;
            // Sum over the 16 h-columns of this tile (within each lane-half)
#pragma unroll
            for (int m = 1; m <= 8; m <<= 1) {
#pragma unroll
                for (int e = 0; e < 8; ++e) p[e] += __shfl_xor(p[e], m, 32);
            }
            if (lh == 0) {
#pragma unroll
                for (int e = 0; e < 8; ++e) part[nt * 16 + hi * 8 + e][ht] = p[e];
            }
        }
    }
    __syncthreads();

    // Deterministic fixed-order reduction over the 32 h-tiles
    if (tid < ROWS) {
        float s = 0.f;
#pragma unroll
        for (int h = 0; h < HTILES; ++h) s += part[tid][h];
        int n = n0 + tid;
        if (n < N) scores[n] = s + b2[0];
    }
}

// ---------------------------------------------------------------------------
// Top-k via exact global bitonic sort of 64-bit keys:
//   key = (~sortable(score) << 32) | index  -> ascending key == descending
//   score with ties broken by lowest index (matches jax.lax.top_k)
// ---------------------------------------------------------------------------
__global__ void init_keys_kernel(const float* __restrict__ scores,
                                 unsigned long long* __restrict__ keys, int N, int total) {
    int i = blockIdx.x * 256 + threadIdx.x;
    if (i >= total) return;
    unsigned long long key;
    if (i < N) {
        unsigned u = __float_as_uint(scores[i]);
        u = (u & 0x80000000u) ? ~u : (u | 0x80000000u);       // monotone f32->u32
        key = ((unsigned long long)(~u) << 32) | (unsigned)i; // descending score
    } else {
        key = 0xFFFFFFFFFFFFFFFFull;                          // padding sorts last
    }
    keys[i] = key;
}

__global__ void bitonic_step_kernel(unsigned long long* __restrict__ keys, int j, int k, int total) {
    int i = blockIdx.x * 256 + threadIdx.x;
    if (i >= total) return;
    int l = i ^ j;
    if (l > i) {
        unsigned long long a = keys[i], b = keys[l];
        bool up = ((i & k) == 0);
        if ((a > b) == up) { keys[i] = b; keys[l] = a; }
    }
}

// ---------------------------------------------------------------------------
// Softmax over the k selected scores (global denominator cancels after
// renormalization, so only top-k values participate). Single block,
// deterministic tree reduction.
// ---------------------------------------------------------------------------
__global__ __launch_bounds__(1024)
void softmax_topk_kernel(const unsigned long long* __restrict__ keys,
                         const float* __restrict__ scores,
                         float* __restrict__ ak_out, float* __restrict__ ak_ws,
                         int* __restrict__ idx_ws, int k) {
    __shared__ float red[1024];
    __shared__ float smax_s;
    int tid = threadIdx.x;
    if (tid == 0) smax_s = scores[(unsigned)(keys[0] & 0xFFFFFFFFu)]; // sorted: key 0 = max
    __syncthreads();
    float smax  = smax_s;
    float local = 0.f;
    for (int i = tid; i < k; i += 1024) {
        int idx = (int)(unsigned)(keys[i] & 0xFFFFFFFFu);
        float e = expf(scores[idx] - smax);
        ak_ws[i]  = e;
        idx_ws[i] = idx;
        local += e;
    }
    red[tid] = local;
    __syncthreads();
    for (int s = 512; s > 0; s >>= 1) {
        if (tid < s) red[tid] += red[tid + s];
        __syncthreads();
    }
    float inv = 1.0f / red[0];
    for (int i = tid; i < k; i += 1024) {
        float a = ak_ws[i] * inv;
        ak_out[i] = a;   // output Ak, in descending-score order
        ak_ws[i]  = a;   // normalized weights for the M kernel
    }
}

// ---------------------------------------------------------------------------
// M[d] = sum_i Ak[i] * x[idx[i], d] : deterministic two-stage reduction
// ---------------------------------------------------------------------------
__global__ void wsum_partial_kernel(const float* __restrict__ x, const float* __restrict__ ak,
                                    const int* __restrict__ idx, float* __restrict__ partial, int k) {
    int d = blockIdx.x * 128 + threadIdx.x;        // gridDim.x = D/128
    int p = blockIdx.y;                            // gridDim.y = 16
    int chunk = (k + 15) / 16;
    int i0 = p * chunk;
    int i1 = i0 + chunk; if (i1 > k) i1 = k;
    float acc = 0.f;
    for (int i = i0; i < i1; ++i)
        acc = fmaf(ak[i], x[(size_t)idx[i] * D_DIM + d], acc);
    partial[p * D_DIM + d] = acc;
}

__global__ void reduce_m_kernel(const float* __restrict__ partial, float* __restrict__ M) {
    int d = blockIdx.x * 256 + threadIdx.x;
    if (d >= D_DIM) return;
    float acc = 0.f;
#pragma unroll
    for (int p = 0; p < 16; ++p) acc += partial[p * D_DIM + d];
    M[d] = acc;
}

// ---------------------------------------------------------------------------
extern "C" void kernel_launch(void* const* d_in, const int* in_sizes, int n_in,
                              void* d_out, int out_size, void* d_ws, size_t ws_size,
                              hipStream_t stream) {
    const float* x  = (const float*)d_in[0];
    const float* W1 = (const float*)d_in[1];
    const float* b1 = (const float*)d_in[2];
    const float* W2 = (const float*)d_in[3];
    const float* b2 = (const float*)d_in[4];
    float* out = (float*)d_out;

    const int D = D_DIM, H = H_DIM;
    const int N = in_sizes[0] / D;                 // 50000
    int k = N / 10; if (k < 1) k = 1;              // 5000 (static)
    int total = 1; while (total < N) total <<= 1;  // 65536

    // Workspace layout (512B aligned chunks)
    char* ws = (char*)d_ws;
    size_t off = 0;
    __bf16* W1bf = (__bf16*)(ws + off);            off += (size_t)H * D * 2;        off = (off + 511) & ~511ull;
    float* scores = (float*)(ws + off);            off += (size_t)N * 4;            off = (off + 511) & ~511ull;
    unsigned long long* keys =
        (unsigned long long*)(ws + off);           off += (size_t)total * 8;        off = (off + 511) & ~511ull;
    float* ak_ws = (float*)(ws + off);             off += (size_t)k * 4;            off = (off + 511) & ~511ull;
    int*   idx_ws = (int*)(ws + off);              off += (size_t)k * 4;            off = (off + 511) & ~511ull;
    float* partial = (float*)(ws + off);           // 16 * D * 4

    // 1) W1 -> bf16 (L2-resident operand for WMMA)
    cvt_w1_kernel<<<(H * D + 255) / 256, 256, 0, stream>>>(W1, W1bf, H * D);

    // 2) Fused MLP scores (the WMMA GEMM)
    scores_kernel<<<(N + ROWS - 1) / ROWS, THREADS, 0, stream>>>(x, W1bf, b1, W2, b2, scores, N);

    // 3) Exact top-k ordering: global bitonic sort of (score,index) keys
    init_keys_kernel<<<(total + 255) / 256, 256, 0, stream>>>(scores, keys, N, total);
    for (int kk = 2; kk <= total; kk <<= 1)
        for (int j = kk >> 1; j > 0; j >>= 1)
            bitonic_step_kernel<<<(total + 255) / 256, 256, 0, stream>>>(keys, j, kk, total);

    // 4) Softmax over top-k (renormalized) -> Ak output + weights
    softmax_topk_kernel<<<1, 1024, 0, stream>>>(keys, scores, out + D, ak_ws, idx_ws, k);

    // 5) M = sum Ak * x[idx]  (deterministic two-stage reduction)
    wsum_partial_kernel<<<dim3(D / 128, 16), 128, 0, stream>>>(x, ak_ws, idx_ws, partial, k);
    reduce_m_kernel<<<(D + 255) / 256, 256, 0, stream>>>(partial, out);
}
